// AFTFull_16492674416918
// MI455X (gfx1250) — matrix operand order repaired and checked by reading, not compiled, not run on hardware
//
#include <hip/hip_runtime.h>

// ---------------------------------------------------------------------------
// AFT-Full on MI455X (gfx1250, wave32, WMMA f16 16x16x32, f32 accumulate)
//
// B=8, NQ=NKV=2048, DQ=DKV=1024, H=256
//
//   0) wqT/wkT/wvT/f2T = transpose(w) as f16 [N][K]   (one-time, 1.7 MiB)
//   1) qsigT[b][h][s]  = sigmoid(x_q @ wq + b)        f16 [B][H][NQ]
//   2) kvcatT[b][h][t]       = exp(x_kv @ wk + b)     f16 [B][2H][NKV] (den)
//   3) kvcatT[b][H+h][t]     = den * (x_kv @ wv + b)  (num)
//   4) yt[b][s][h] = qsigT * (expW @ num) / (expW @ den)   f16 [16384, 256]
//   5) out = yt @ f2 + f2_b                           f32 [16384, 256]
//
// LDS staging is b128 except the per-half f32->f16 A-tile writes (transposes live in global layouts, not
// in the kernel); global loads are software-pipelined past the WMMA phase.
// Workspace: ~34 MiB of d_ws.
// ---------------------------------------------------------------------------

typedef __attribute__((ext_vector_type(16))) _Float16 v16h;
typedef __attribute__((ext_vector_type(8)))  float    v8f;
typedef __attribute__((ext_vector_type(4)))  float    v4f;
typedef __attribute__((ext_vector_type(4)))  unsigned int v4u;

#define BLOCK 256      // 8 waves (wave32)
#define TM 64
#define TN 128
#define TK 32
#define LDA_SH 40      // halves/row: 80B rows -> 16B aligned, 20-bank stride
#define LDB_SH 40

enum EpiKind { EPI_SIG16T = 0, EPI_EXP16T = 1, EPI_MULAUXT = 2, EPI_BIAS32 = 3 };

union H8 { v4u v; _Float16 h[8]; };

// A fragment (16x32 f16). ISA layout: lane L holds row M=L%16; halves
// {8hi+2j} = dwords 4hi..4hi+3 and {16+8hi+2j} = dwords 8+4hi..11+4hi,
// each a 16B chunk -> two ds_load_b128.
__device__ __forceinline__ v16h ld_a_frag(const _Float16* As, int mbase, int lane) {
  const int hi  = lane >> 4;
  const int row = mbase + (lane & 15);
  const v4u* p = (const v4u*)(As + row * LDA_SH);
  union { v16h v; v4u q[2]; } f;
  f.q[0] = p[hi];
  f.q[1] = p[2 + hi];
  return f.v;
}

// B fragment (32x16 f16): N = lane%16, K = 8*hi..8*hi+7 and 16+8*hi..16+8*hi+7.
// LDS tile stored [n][k] -> two 16B chunks per lane.
__device__ __forceinline__ v16h ld_b_frag(const _Float16* Bs, int nbase, int lane) {
  const int hi = lane >> 4;
  const int n  = nbase + (lane & 15);
  const v4u* p = (const v4u*)(Bs + n * LDB_SH);
  union { v16h v; v4u q[2]; } f;
  f.q[0] = p[hi];
  f.q[1] = p[2 + hi];
  return f.v;
}

__device__ __forceinline__ v8f wmma_f16(v16h a, v16h b, v8f c) {
  return __builtin_amdgcn_wmma_f32_16x16x32_f16(false, a, false, b, (short)0, c,
                                                false, false);
}

// ---------------------------------------------------------------------------
// One-time weight transpose: W f32 [K][N] -> WT f16 [N][K]
// ---------------------------------------------------------------------------
__global__ __launch_bounds__(BLOCK) void transpose_w_kernel(
    const float* __restrict__ W, _Float16* __restrict__ WT, int K, int N) {
  const int idx = blockIdx.x * BLOCK + threadIdx.x;
  if (idx < K * N) {
    const int k = idx / N, n = idx - k * N;
    WT[(size_t)n * K + k] = (_Float16)W[idx];
  }
}

// ---------------------------------------------------------------------------
// Generic GEMM: out = epi(A[M,K] @ B[K,N] + bias[N]).
// A: f32 or f16 row-major. BT: f16 pre-transposed [N][K].
// T-epilogues write transposed per-batch layouts [N][rpb] (one b128 store per
// lane per 16x16 quadrant).
// ---------------------------------------------------------------------------
template <typename AT, typename OT, int EPI>
__global__ __launch_bounds__(BLOCK) void gemm_kernel(
    const AT* __restrict__ A, int lda,
    const _Float16* __restrict__ BT,       // [N][K]
    const float* __restrict__ bias,
    OT* __restrict__ out, int ldo,
    const _Float16* __restrict__ aux, int ldaux,
    int K, int rpb, size_t bstride) {
  __shared__ __align__(16) _Float16 As[TM * LDA_SH];
  __shared__ __align__(16) _Float16 Bs[TN * LDB_SH];

  const int m0 = blockIdx.x * TM, n0 = blockIdx.y * TN;
  const int tid = threadIdx.x, lane = tid & 31, wid = tid >> 5;
  const int wm = (wid & 1) * 32, wn = (wid >> 1) * 32;
  const bool A32 = (sizeof(AT) == 4);

  v4f aF[2];   // A f32 regs: it=tid+256j -> r=it>>3, c=(it&7)*4
  v4u aH;      // A f16 regs: r=tid>>2, c=(tid&3)*8
  v4u bH[2];   // B regs:     it=tid+256j -> n=it>>2, k8=(it&3)*8

  auto g_load = [&](int k0) {
    if (A32) {
#pragma unroll
      for (int j = 0; j < 2; ++j) {
        const int it = tid + BLOCK * j, r = it >> 3, c = (it & 7) * 4;
        aF[j] = *(const v4f*)((const float*)A + (size_t)(m0 + r) * lda + k0 + c);
      }
    } else {
      const int r = tid >> 2, c = (tid & 3) * 8;
      aH = *(const v4u*)((const _Float16*)A + (size_t)(m0 + r) * lda + k0 + c);
    }
#pragma unroll
    for (int j = 0; j < 2; ++j) {
      const int it = tid + BLOCK * j, n = it >> 2, k8 = (it & 3) * 8;
      bH[j] = *(const v4u*)(BT + (size_t)(n0 + n) * K + k0 + k8);
    }
  };

  auto s_store = [&]() {
    if (A32) {
#pragma unroll
      for (int j = 0; j < 2; ++j) {
        const int it = tid + BLOCK * j, r = it >> 3, c = (it & 7) * 4;
#pragma unroll
        for (int q = 0; q < 4; ++q) As[r * LDA_SH + c + q] = (_Float16)aF[j][q];
      }
    } else {
      const int r = tid >> 2, c = (tid & 3) * 8;
      *(v4u*)(As + r * LDA_SH + c) = aH;
    }
#pragma unroll
    for (int j = 0; j < 2; ++j) {
      const int it = tid + BLOCK * j, n = it >> 2, k8 = (it & 3) * 8;
      *(v4u*)(Bs + n * LDB_SH + k8) = bH[j];
    }
  };

  v8f acc[2][2] = {};
  g_load(0);
  for (int k0 = 0; k0 < K; k0 += TK) {
    __syncthreads();
    s_store();
    __syncthreads();
    if (k0 + TK < K) g_load(k0 + TK);  // overlaps with frag reads + WMMA

    const v16h a0 = ld_a_frag(As, wm,      lane);
    const v16h a1 = ld_a_frag(As, wm + 16, lane);
    const v16h b0 = ld_b_frag(Bs, wn,      lane);
    const v16h b1 = ld_b_frag(Bs, wn + 16, lane);
    acc[0][0] = wmma_f16(a0, b0, acc[0][0]);
    acc[0][1] = wmma_f16(a0, b1, acc[0][1]);
    acc[1][0] = wmma_f16(a1, b0, acc[1][0]);
    acc[1][1] = wmma_f16(a1, b1, acc[1][1]);
  }

  // C/D layout: lane L, VGPR r -> M = r + 8*(L>=16), N = L%16.
  const int hi  = lane >> 4;
  const int nlo = lane & 15;
  if (EPI == EPI_BIAS32) {
#pragma unroll
    for (int ti = 0; ti < 2; ++ti)
#pragma unroll
      for (int tj = 0; tj < 2; ++tj)
#pragma unroll
        for (int r = 0; r < 8; ++r) {
          const int m = m0 + wm + 16 * ti + 8 * hi + r;
          const int n = n0 + wn + 16 * tj + nlo;
          out[(size_t)m * ldo + n] = (OT)(acc[ti][tj][r] + bias[n]);
        }
  } else {
    const int bb = m0 / rpb;                 // batch of this tile
    OT* outB = out + (size_t)bb * bstride;
    const _Float16* auxB = aux + (size_t)bb * bstride;
#pragma unroll
    for (int ti = 0; ti < 2; ++ti)
#pragma unroll
      for (int tj = 0; tj < 2; ++tj) {
        const int ml = (m0 - bb * rpb) + wm + 16 * ti + 8 * hi;  // row-in-batch
        const int n  = n0 + wn + 16 * tj + nlo;
        const float bn = bias[n];
        H8 o, ax;
        if (EPI == EPI_MULAUXT) ax.v = *(const v4u*)(auxB + (size_t)n * ldaux + ml);
#pragma unroll
        for (int r = 0; r < 8; ++r) {
          const float v = acc[ti][tj][r] + bn;
          if (EPI == EPI_SIG16T)       o.h[r] = (_Float16)(1.0f / (1.0f + __expf(-v)));
          else if (EPI == EPI_EXP16T)  o.h[r] = (_Float16)__expf(v);
          else                         o.h[r] = (_Float16)((float)ax.h[r] * v);
        }
        *(v4u*)(outB + (size_t)n * ldo + ml) = o.v;  // 16B store along M
      }
  }
}

// ---------------------------------------------------------------------------
// Core: exp(bias) A-operand feeds BOTH num and den accumulators (8 WMMAs per
// k-step from one A pair). B tiles come from kvcatT [B][2H][NKV] with pure
// b128 copies (no transpose). Epilogue emits Yt f16 [B*NQ][H].
// ---------------------------------------------------------------------------
__global__ __launch_bounds__(BLOCK) void aft_core_kernel(
    const float* __restrict__ bias,      // [B, NQ, NKV]
    const _Float16* __restrict__ kvcatT, // [B][2H][NKV]: rows h=den, H+h=num
    const _Float16* __restrict__ qsigT,  // [B][H][NQ]
    _Float16* __restrict__ yt,           // [B*NQ][H]
    int NQ, int NKV, int H) {
  __shared__ __align__(16) _Float16 As[TM * LDA_SH];
  __shared__ __align__(16) _Float16 Bd[TN * LDB_SH];
  __shared__ __align__(16) _Float16 Bn[TN * LDB_SH];

  const int b  = blockIdx.z;
  const int m0 = blockIdx.x * TM, n0 = blockIdx.y * TN;
  const int tid = threadIdx.x, lane = tid & 31, wid = tid >> 5;
  const int wm = (wid & 1) * 32, wn = (wid >> 1) * 32;

  const float*    biasB = bias   + (size_t)b * NQ * NKV;
  const _Float16* kvTB  = kvcatT + (size_t)b * (2 * H) * NKV;

  v4f aF[2];        // bias tile:  it=tid+256j -> r=it>>3, c=(it&7)*4
  v4u dH[2], nH[2]; // kvT tiles:  it=tid+256j -> n=it>>2, k8=(it&3)*8

  auto g_load = [&](int k0) {
#pragma unroll
    for (int j = 0; j < 2; ++j) {
      const int it = tid + BLOCK * j, r = it >> 3, c = (it & 7) * 4;
      aF[j] = *(const v4f*)(biasB + (size_t)(m0 + r) * NKV + k0 + c);
    }
#pragma unroll
    for (int j = 0; j < 2; ++j) {
      const int it = tid + BLOCK * j, n = it >> 2, k8 = (it & 3) * 8;
      const _Float16* src = kvTB + (size_t)(n0 + n) * NKV + k0 + k8;
      dH[j] = *(const v4u*)(src);                      // den row h
      nH[j] = *(const v4u*)(src + (size_t)H * NKV);    // num row H+h
    }
  };

  auto s_store = [&]() {
#pragma unroll
    for (int j = 0; j < 2; ++j) {
      const int it = tid + BLOCK * j, r = it >> 3, c = (it & 7) * 4;
#pragma unroll
      for (int q = 0; q < 4; ++q)
        As[r * LDA_SH + c + q] = (_Float16)__expf(aF[j][q]);  // fused exp
    }
#pragma unroll
    for (int j = 0; j < 2; ++j) {
      const int it = tid + BLOCK * j, n = it >> 2, k8 = (it & 3) * 8;
      *(v4u*)(Bd + n * LDB_SH + k8) = dH[j];
      *(v4u*)(Bn + n * LDB_SH + k8) = nH[j];
    }
  };

  v8f den[2][2] = {};
  v8f num[2][2] = {};
  g_load(0);
  for (int k0 = 0; k0 < NKV; k0 += TK) {
    __syncthreads();
    s_store();
    __syncthreads();
    if (k0 + TK < NKV) g_load(k0 + TK);

    const v16h a0  = ld_a_frag(As, wm,      lane);
    const v16h a1  = ld_a_frag(As, wm + 16, lane);
    const v16h bd0 = ld_b_frag(Bd, wn,      lane);
    const v16h bd1 = ld_b_frag(Bd, wn + 16, lane);
    const v16h bn0 = ld_b_frag(Bn, wn,      lane);
    const v16h bn1 = ld_b_frag(Bn, wn + 16, lane);
    den[0][0] = wmma_f16(a0, bd0, den[0][0]);
    den[0][1] = wmma_f16(a0, bd1, den[0][1]);
    den[1][0] = wmma_f16(a1, bd0, den[1][0]);
    den[1][1] = wmma_f16(a1, bd1, den[1][1]);
    num[0][0] = wmma_f16(a0, bn0, num[0][0]);
    num[0][1] = wmma_f16(a0, bn1, num[0][1]);
    num[1][0] = wmma_f16(a1, bn0, num[1][0]);
    num[1][1] = wmma_f16(a1, bn1, num[1][1]);
  }

  const int hi  = lane >> 4;
  const int nlo = lane & 15;
  const _Float16* qTB = qsigT + (size_t)b * H * NQ;
  _Float16*       yB  = yt    + (size_t)b * NQ * H;
#pragma unroll
  for (int ti = 0; ti < 2; ++ti)
#pragma unroll
    for (int tj = 0; tj < 2; ++tj) {
      const int mb = m0 + wm + 16 * ti + 8 * hi;
      const int n  = n0 + wn + 16 * tj + nlo;
      H8 qv;
      qv.v = *(const v4u*)(qTB + (size_t)n * NQ + mb);  // b128 gather along M
#pragma unroll
      for (int r = 0; r < 8; ++r) {
        const float qs = (float)qv.h[r];
        const float u  = num[ti][tj][r];
        const float d  = den[ti][tj][r];
        yB[(size_t)(mb + r) * H + n] = (_Float16)(qs * (u / d));
      }
    }
}

// ---------------------------------------------------------------------------

extern "C" void kernel_launch(void* const* d_in, const int* in_sizes, int n_in,
                              void* d_out, int out_size, void* d_ws, size_t ws_size,
                              hipStream_t stream) {
  (void)in_sizes; (void)n_in; (void)out_size; (void)ws_size;

  const float* x_q  = (const float*)d_in[0];
  const float* x_kv = (const float*)d_in[1];
  const float* bias = (const float*)d_in[2];
  const float* wq_w = (const float*)d_in[3];
  const float* wq_b = (const float*)d_in[4];
  const float* wk_w = (const float*)d_in[5];
  const float* wk_b = (const float*)d_in[6];
  const float* wv_w = (const float*)d_in[7];
  const float* wv_b = (const float*)d_in[8];
  const float* f2_w = (const float*)d_in[9];
  const float* f2_b = (const float*)d_in[10];
  float* out = (float*)d_out;

  const int B = 8, NQ = 2048, NKV = 2048, DQ = 1024, DKV = 1024, H = 256;
  const int Mq  = B * NQ;    // 16384
  const int Mkv = B * NKV;   // 16384

  char* ws = (char*)d_ws;
  _Float16* qsigT  = (_Float16*)ws;  ws += (size_t)B * H * NQ * sizeof(_Float16);      // 8 MiB
  _Float16* kvcatT = (_Float16*)ws;  ws += (size_t)B * 2 * H * NKV * sizeof(_Float16); // 16 MiB
  _Float16* yt     = (_Float16*)ws;  ws += (size_t)Mq * H * sizeof(_Float16);          // 8 MiB
  _Float16* wqT    = (_Float16*)ws;  ws += (size_t)DQ * H * sizeof(_Float16);
  _Float16* wkT    = (_Float16*)ws;  ws += (size_t)DKV * H * sizeof(_Float16);
  _Float16* wvT    = (_Float16*)ws;  ws += (size_t)DKV * H * sizeof(_Float16);
  _Float16* f2T    = (_Float16*)ws;

  const dim3 blk(BLOCK);

  // 0) one-time weight transposes (f32 [K][N] -> f16 [N][K])
  transpose_w_kernel<<<(DQ * H + BLOCK - 1) / BLOCK, blk, 0, stream>>>(wq_w, wqT, DQ, H);
  transpose_w_kernel<<<(DKV * H + BLOCK - 1) / BLOCK, blk, 0, stream>>>(wk_w, wkT, DKV, H);
  transpose_w_kernel<<<(DKV * H + BLOCK - 1) / BLOCK, blk, 0, stream>>>(wv_w, wvT, DKV, H);
  transpose_w_kernel<<<(H * H + BLOCK - 1) / BLOCK, blk, 0, stream>>>(f2_w, f2T, H, H);

  // 1) qsigT[b][h][s] = sigmoid(x_q @ wq + b)
  gemm_kernel<float, _Float16, EPI_SIG16T>
      <<<dim3(Mq / TM, H / TN), blk, 0, stream>>>(
          x_q, DQ, wqT, wq_b, qsigT, NQ, nullptr, 0, DQ, NQ, (size_t)H * NQ);

  // 2) kvcatT[b][h][t] = exp(x_kv @ wk + b)        (den rows)
  gemm_kernel<float, _Float16, EPI_EXP16T>
      <<<dim3(Mkv / TM, H / TN), blk, 0, stream>>>(
          x_kv, DKV, wkT, wk_b, kvcatT, NKV, nullptr, 0, DKV, NKV,
          (size_t)2 * H * NKV);

  // 3) kvcatT[b][H+h][t] = den * (x_kv @ wv + b)   (num rows)
  gemm_kernel<float, _Float16, EPI_MULAUXT>
      <<<dim3(Mkv / TM, H / TN), blk, 0, stream>>>(
          x_kv, DKV, wvT, wv_b, kvcatT + (size_t)H * NKV, NKV, kvcatT, NKV, DKV,
          NKV, (size_t)2 * H * NKV);

  // 4) yt = qsig * (exp(bias) @ num) / (exp(bias) @ den)
  aft_core_kernel<<<dim3(NQ / TM, H / TN, B), blk, 0, stream>>>(
      bias, kvcatT, qsigT, yt, NQ, NKV, H);

  // 5) out = yt @ f2 + f2_b
  gemm_kernel<_Float16, float, EPI_BIAS32>
      <<<dim3(Mq / TM, H / TN), blk, 0, stream>>>(
          yt, H, f2T, f2_b, out, H, nullptr, 0, H, NQ, 0);
}
